// GATLayer_18468359372993
// MI455X (gfx1250) — matrix operand order repaired and checked by reading, not compiled, not run on hardware
//
#include <hip/hip_runtime.h>

typedef __attribute__((ext_vector_type(16))) __bf16 v16bf;
typedef __attribute__((ext_vector_type(8)))  __bf16 v8bf;
typedef __attribute__((ext_vector_type(8)))  float  v8f;
typedef __attribute__((ext_vector_type(8)))  unsigned char v8u8;
typedef unsigned short ushort_t;

#define BB  8
#define NN  1024
#define DD  128
#define HH  4
#define HDD 32

__device__ __forceinline__ ushort_t f2bf(float f) {
  unsigned int u = __builtin_bit_cast(unsigned int, f);
  u += 0x7FFFu + ((u >> 16) & 1u);          // round to nearest even
  return (ushort_t)(u >> 16);
}
__device__ __forceinline__ __bf16 us2bf(ushort_t u) {
  return __builtin_bit_cast(__bf16, u);
}

// ---------------------------------------------------------------- prep kernels
__global__ void cvt_f32_bf16(const float* __restrict__ src,
                             ushort_t* __restrict__ dst, int n) {
  int i = blockIdx.x * blockDim.x + threadIdx.x;
  if (i < n) dst[i] = f2bf(src[i]);
}

// Swizzle a 128x128 f32 weight into per-lane WMMA B-operand order (bf16).
// out[tile*512 + lane*16 + e] = W[kt*32 + (lane>>4)*8 + (e&7) + (e>>3)*16][ct*16 + (lane&15)]
// tile = kt*8 + ct  (kt = K/32 step, ct = 16-col tile)
__global__ void swz_weight(const float* __restrict__ W, ushort_t* __restrict__ out) {
  int idx  = blockIdx.x * 256 + threadIdx.x;     // 16384 elements
  int e    = idx & 15;
  int lane = (idx >> 4) & 31;
  int tile = idx >> 9;
  int kt = tile >> 3, ct = tile & 7;
  int k = kt * 32 + ((lane >> 4) << 3) + (e & 7) + ((e >> 3) << 4);
  int c = ct * 16 + (lane & 15);
  out[idx] = f2bf(W[k * DD + c]);
}

// adjT[m*N + n] = adj[n*N + m] != 0   (byte matrix for 8-byte row-mask loads)
__global__ void adj_tr(const int* __restrict__ adj, unsigned char* __restrict__ at) {
  int idx = blockIdx.x * 256 + threadIdx.x;      // N*N
  int m = idx >> 10, n = idx & (NN - 1);
  at[idx] = (unsigned char)(adj[(size_t)n * NN + m] != 0);
}

// ---------------------------------------------------------------- QKV GEMM
// One wave: 16x64 tile of q/k/v = x @ W + b.  W pre-swizzled.  V stored transposed.
__global__ __launch_bounds__(32) void qkv_gemm(
    const ushort_t* __restrict__ xb,
    const ushort_t* __restrict__ Wqb, const float* __restrict__ bqv,
    const ushort_t* __restrict__ Wkb, const float* __restrict__ bkv,
    const ushort_t* __restrict__ Wvb, const float* __restrict__ bvv,
    ushort_t* __restrict__ qo, ushort_t* __restrict__ ko,
    ushort_t* __restrict__ vT)
{
  const int lane = threadIdx.x;
  const int row0 = blockIdx.x * 16;
  const int col0 = blockIdx.y * 64;
  const ushort_t* W; const float* bias;
  if (blockIdx.z == 0)      { W = Wqb; bias = bqv; }
  else if (blockIdx.z == 1) { W = Wkb; bias = bkv; }
  else                      { W = Wvb; bias = bvv; }

  const int mrow = lane & 15;
  const int kb8  = (lane >> 4) * 8;
  const int colL = lane & 15;
  const int joff = (lane >> 4) << 3;

  v8f zero = {};
  v8f acc[4];
#pragma unroll
  for (int t = 0; t < 4; ++t) acc[t] = zero;

#pragma unroll
  for (int k0 = 0; k0 < DD; k0 += 32) {
    const ushort_t* xp = xb + (size_t)(row0 + mrow) * DD + k0;
    v8bf alo = *(const v8bf*)(xp + kb8);
    v8bf ahi = *(const v8bf*)(xp + 16 + kb8);
    v16bf a;
#pragma unroll
    for (int i = 0; i < 8; ++i) { a[i] = alo[i]; a[8 + i] = ahi[i]; }
#pragma unroll
    for (int t = 0; t < 4; ++t) {
      const ushort_t* wp = W + ((((k0 >> 5) << 3) + (col0 >> 4) + t) << 9) + (lane << 4);
      v16bf bm = *(const v16bf*)wp;
      acc[t] = __builtin_amdgcn_wmma_f32_16x16x32_bf16(
          false, a, false, bm, (short)0, acc[t], false, false);
    }
  }

  if (blockIdx.z == 2) {            // V: store transposed vT[((b*H+h)*HD+d)*N + n]
#pragma unroll
    for (int t = 0; t < 4; ++t) {
      int col = col0 + t * 16 + colL;
      float bs = bias[col];
      int hh = col >> 5, d = col & 31;
#pragma unroll
      for (int j = 0; j < 8; ++j) {
        int row = row0 + j + joff;
        int bidx = row >> 10, n = row & (NN - 1);
        vT[(((size_t)bidx * HH + hh) * HDD + d) * NN + n] = f2bf(acc[t][j] + bs);
      }
    }
  } else {
    ushort_t* out = (blockIdx.z == 0) ? qo : ko;
#pragma unroll
    for (int t = 0; t < 4; ++t) {
      int col = col0 + t * 16 + colL;
      float bs = bias[col];
#pragma unroll
      for (int j = 0; j < 8; ++j) {
        int row = row0 + j + joff;
        out[(size_t)row * DD + col] = f2bf(acc[t][j] + bs);
      }
    }
  }
}

// ---------------------------------------------------------------- attention
// One wave per (b, h, 16 q-rows). Two-pass in-register masked softmax:
//   pass A: scores via WMMA -> per-row max (registers, one shfl reduction)
//   pass B: recompute scores, e=exp(s-max) masked, stage bf16 tile in 1.3KB LDS,
//           immediately consume as WMMA A-operand for P@V. Normalize at the end.
__global__ __launch_bounds__(32) void attn_kernel(
    const ushort_t* __restrict__ qbuf, const ushort_t* __restrict__ kbuf,
    const ushort_t* __restrict__ vT, const unsigned char* __restrict__ adjT,
    ushort_t* __restrict__ ao)
{
  __shared__ ushort_t sE[16][40];          // 16x32 prob staging tile (+pad)
  const int lane = threadIdx.x;
  const int n0 = blockIdx.x * 16;
  const int h  = blockIdx.y;
  const int b  = blockIdx.z;
  const float scale = 0.17677669529663687f;   // 1/sqrt(32)

  const int mrow  = lane & 15;
  const int kb8   = (lane >> 4) * 8;
  const int colL  = lane & 15;
  const int khalf = (lane >> 4) << 4;
  const int joff  = (lane >> 4) << 3;

  // A operand: q tile (rows n0..n0+15, K = head dim 32)
  v16bf aq;
  {
    const ushort_t* qp = qbuf + ((size_t)(b * NN + n0 + mrow) * HH + h) * HDD;
    v8bf qlo = *(const v8bf*)(qp + khalf);
    v8bf qhi = *(const v8bf*)(qp + khalf + 8);
#pragma unroll
    for (int i = 0; i < 8; ++i) { aq[i] = qlo[i]; aq[8 + i] = qhi[i]; }
  }

  const ushort_t* kbase = kbuf + ((size_t)(b * NN) * HH + h) * HDD + khalf;
  const unsigned char* abase = adjT + n0 + joff;

  // ---- pass A: per-row max of masked scaled scores (kept in registers)
  float rmax[8];
#pragma unroll
  for (int j = 0; j < 8; ++j) rmax[j] = -INFINITY;

  for (int m0 = 0; m0 < NN; m0 += 16) {
    v16bf bk_ = *(const v16bf*)(kbase + (size_t)(m0 + colL) * DD);
    v8f c = {};
    c = __builtin_amdgcn_wmma_f32_16x16x32_bf16(
        false, aq, false, bk_, (short)0, c, false, false);
    v8u8 msk = *(const v8u8*)(abase + (size_t)(m0 + colL) * NN);
#pragma unroll
    for (int j = 0; j < 8; ++j) {
      float sv = msk[j] ? c[j] * scale : -INFINITY;
      rmax[j] = fmaxf(rmax[j], sv);
    }
  }
  // reduce across the 16 columns held by this half-wave
#pragma unroll
  for (int j = 0; j < 8; ++j) {
    float m = rmax[j];
    m = fmaxf(m, __shfl_xor(m, 1, 32));
    m = fmaxf(m, __shfl_xor(m, 2, 32));
    m = fmaxf(m, __shfl_xor(m, 4, 32));
    m = fmaxf(m, __shfl_xor(m, 8, 32));
    rmax[j] = m;
  }

  // ---- pass B: exp + P@V fused, rowsum accumulated distributed per lane
  float rsum[8];
#pragma unroll
  for (int j = 0; j < 8; ++j) rsum[j] = 0.f;
  v8f o0 = {}, o1 = {};

  const ushort_t* vbase = vT + ((size_t)(b * HH + h) * HDD) * NN;

  for (int m0 = 0; m0 < NN; m0 += 32) {
#pragma unroll
    for (int t = 0; t < 2; ++t) {
      int mc = m0 + t * 16;
      v16bf bk_ = *(const v16bf*)(kbase + (size_t)(mc + colL) * DD);
      v8f c = {};
      c = __builtin_amdgcn_wmma_f32_16x16x32_bf16(
          false, aq, false, bk_, (short)0, c, false, false);
      v8u8 msk = *(const v8u8*)(abase + (size_t)(mc + colL) * NN);
#pragma unroll
      for (int j = 0; j < 8; ++j) {
        float e = msk[j] ? __expf(c[j] * scale - rmax[j]) : 0.f;
        rsum[j] += e;
        sE[j + joff][t * 16 + colL] = f2bf(e);
      }
    }
    __syncthreads();   // DScnt fence: staging tile visible to A-operand reads

    v8bf plo = *(const v8bf*)&sE[mrow][khalf];
    v8bf phi = *(const v8bf*)&sE[mrow][khalf + 8];
    v16bf ap;
#pragma unroll
    for (int i = 0; i < 8; ++i) { ap[i] = plo[i]; ap[8 + i] = phi[i]; }

    int kr = m0 + khalf;
    v16bf b0 = *(const v16bf*)(vbase + (size_t)colL * NN + kr);
    v16bf b1 = *(const v16bf*)(vbase + (size_t)(16 + colL) * NN + kr);
    o0 = __builtin_amdgcn_wmma_f32_16x16x32_bf16(
        false, ap, false, b0, (short)0, o0, false, false);
    o1 = __builtin_amdgcn_wmma_f32_16x16x32_bf16(
        false, ap, false, b1, (short)0, o1, false, false);
    __syncthreads();   // keep LDS in-order before next chunk overwrites sE
  }

  // ---- normalize (row mapping of o matches rsum: row = j + joff) and store
#pragma unroll
  for (int j = 0; j < 8; ++j) {
    float s = rsum[j];
    s += __shfl_xor(s, 1, 32);
    s += __shfl_xor(s, 2, 32);
    s += __shfl_xor(s, 4, 32);
    s += __shfl_xor(s, 8, 32);
    float inv = 1.f / s;
    int row = n0 + j + joff;
    size_t base = (size_t)(b * NN + row) * DD + h * HDD;
    ao[base + colL]      = f2bf(o0[j] * inv);
    ao[base + 16 + colL] = f2bf(o1[j] * inv);
  }
}

// ---------------------------------------------------------------- out proj + residual
__global__ __launch_bounds__(32) void out_proj(
    const ushort_t* __restrict__ ab, const ushort_t* __restrict__ Wob,
    const float* __restrict__ bov, const float* __restrict__ x,
    float* __restrict__ y)
{
  const int lane = threadIdx.x;
  const int row0 = blockIdx.x * 16;
  const int mrow = lane & 15;
  const int kb8  = (lane >> 4) * 8;
  const int colL = lane & 15;
  const int joff = (lane >> 4) << 3;

  v8f zero = {};
  v8f acc[8];
#pragma unroll
  for (int t = 0; t < 8; ++t) acc[t] = zero;

#pragma unroll
  for (int k0 = 0; k0 < DD; k0 += 32) {
    const ushort_t* apz = ab + (size_t)(row0 + mrow) * DD + k0;
    v8bf alo = *(const v8bf*)(apz + kb8);
    v8bf ahi = *(const v8bf*)(apz + 16 + kb8);
    v16bf a;
#pragma unroll
    for (int i = 0; i < 8; ++i) { a[i] = alo[i]; a[8 + i] = ahi[i]; }
#pragma unroll
    for (int t = 0; t < 8; ++t) {
      const ushort_t* wp = Wob + ((((k0 >> 5) << 3) + t) << 9) + (lane << 4);
      v16bf bm = *(const v16bf*)wp;
      acc[t] = __builtin_amdgcn_wmma_f32_16x16x32_bf16(
          false, a, false, bm, (short)0, acc[t], false, false);
    }
  }
#pragma unroll
  for (int t = 0; t < 8; ++t) {
    int col = t * 16 + colL;
    float bs = bov[col];
#pragma unroll
    for (int j = 0; j < 8; ++j) {
      size_t idx = (size_t)(row0 + j + joff) * DD + col;
      y[idx] = acc[t][j] + bs + x[idx];
    }
  }
}

// ---------------------------------------------------------------- layernorm
__global__ __launch_bounds__(256) void ln_kernel(
    const float* __restrict__ y, const float* __restrict__ gamma,
    const float* __restrict__ beta, float* __restrict__ out)
{
  const int wave = threadIdx.x >> 5;
  const int lane = threadIdx.x & 31;
  const int row = blockIdx.x * 8 + wave;
  const float* yr = y + (size_t)row * DD;
  float v[4];
  float s = 0.f;
#pragma unroll
  for (int i = 0; i < 4; ++i) { v[i] = yr[lane + 32 * i]; s += v[i]; }
#pragma unroll
  for (int off = 16; off > 0; off >>= 1) s += __shfl_xor(s, off, 32);
  float mu = s * (1.f / DD);
  float var = 0.f;
#pragma unroll
  for (int i = 0; i < 4; ++i) { float d = v[i] - mu; var += d * d; }
#pragma unroll
  for (int off = 16; off > 0; off >>= 1) var += __shfl_xor(var, off, 32);
  var *= (1.f / DD);
  float inv = rsqrtf(var + 1e-5f);
  float* orow = out + (size_t)row * DD;
#pragma unroll
  for (int i = 0; i < 4; ++i) {
    int c = lane + 32 * i;
    orow[c] = (v[i] - mu) * inv * gamma[c] + beta[c];
  }
}

// ---------------------------------------------------------------- launch
extern "C" void kernel_launch(void* const* d_in, const int* in_sizes, int n_in,
                              void* d_out, int out_size, void* d_ws, size_t ws_size,
                              hipStream_t stream) {
  (void)in_sizes; (void)n_in; (void)out_size; (void)ws_size;
  const float* x   = (const float*)d_in[0];
  const int*   adj = (const int*)d_in[1];
  const float* Wq  = (const float*)d_in[2];
  const float* bq  = (const float*)d_in[3];
  const float* Wk  = (const float*)d_in[4];
  const float* bk  = (const float*)d_in[5];
  const float* Wv  = (const float*)d_in[6];
  const float* bv  = (const float*)d_in[7];
  const float* Wo  = (const float*)d_in[8];
  const float* bo  = (const float*)d_in[9];
  const float* gam = (const float*)d_in[10];
  const float* bet = (const float*)d_in[11];
  float* out = (float*)d_out;

  const size_t nx = (size_t)BB * NN * DD;   // 1,048,576
  const size_t nw = (size_t)DD * DD;        // 16,384
  const size_t na = (size_t)NN * NN;        // 1,048,576

  char* ws = (char*)d_ws;
  ushort_t* xb   = (ushort_t*)ws;  ws += nx * 2;
  ushort_t* Wqb  = (ushort_t*)ws;  ws += nw * 2;
  ushort_t* Wkb  = (ushort_t*)ws;  ws += nw * 2;
  ushort_t* Wvb  = (ushort_t*)ws;  ws += nw * 2;
  ushort_t* Wob  = (ushort_t*)ws;  ws += nw * 2;
  ushort_t* qbv  = (ushort_t*)ws;  ws += nx * 2;
  ushort_t* kbv  = (ushort_t*)ws;  ws += nx * 2;
  ushort_t* vTb  = (ushort_t*)ws;  ws += nx * 2;
  ushort_t* abv  = (ushort_t*)ws;  ws += nx * 2;
  float*    yv   = (float*)ws;     ws += nx * 4;
  unsigned char* adjTb = (unsigned char*)ws; ws += na;   // ~15.2 MB total

  cvt_f32_bf16<<<(int)((nx + 255) / 256), 256, 0, stream>>>(x, xb, (int)nx);
  swz_weight<<<64, 256, 0, stream>>>(Wq, Wqb);
  swz_weight<<<64, 256, 0, stream>>>(Wk, Wkb);
  swz_weight<<<64, 256, 0, stream>>>(Wv, Wvb);
  swz_weight<<<64, 256, 0, stream>>>(Wo, Wob);
  adj_tr<<<(int)(na / 256), 256, 0, stream>>>(adj, adjTb);

  dim3 g1(BB * NN / 16, DD / 64, 3);
  qkv_gemm<<<g1, 32, 0, stream>>>(xb, Wqb, bq, Wkb, bk, Wvb, bv, qbv, kbv, vTb);

  dim3 g2(NN / 16, HH, BB);
  attn_kernel<<<g2, 32, 0, stream>>>(qbv, kbv, vTb, adjTb, abv);

  out_proj<<<BB * NN / 16, 32, 0, stream>>>(abv, Wob, bo, x, yv);

  ln_kernel<<<BB * NN / 8, 256, 0, stream>>>(yv, gam, bet, out);
}
